// InformerEncoder_23682449670173
// MI455X (gfx1250) — matrix operand order repaired and checked by reading, not compiled, not run on hardware
//
#include <hip/hip_runtime.h>
#include <math.h>

// ---------------------------------------------------------------------------
// Informer encoder forward for MI455X (gfx1250), wave32 + WMMA.
// Heavy GEMMs: A panel staged to LDS via async-to-LDS loads (ASYNCcnt path),
// f16 N-major weights via b128 loads -> v_wmma_f32_16x16x32_f16 -> f32 acc.
// ---------------------------------------------------------------------------

typedef __attribute__((ext_vector_type(16))) _Float16 v16h;
typedef __attribute__((ext_vector_type(8)))  float    v8f;
typedef __attribute__((ext_vector_type(4)))  float    v4f;
typedef __attribute__((ext_vector_type(4)))  int      v4i;

#define BATCH     64
#define DMODEL    512
#define NHEADS    8
#define DH        64
#define DFF       2048

#if __has_builtin(__builtin_amdgcn_global_load_async_to_lds_b128) && \
    __has_builtin(__builtin_amdgcn_s_wait_asynccnt)
#define HAS_ASYNC_LDS 1
typedef __attribute__((address_space(1))) v4i* gptr4;   // global int4*
typedef __attribute__((address_space(3))) v4i* sptr4;   // LDS int4*
#else
#define HAS_ASYNC_LDS 0
#endif

__device__ __forceinline__ float gelu_exact(float x) {
    return 0.5f * x * (1.0f + erff(x * 0.70710678118654752440f));
}

// ---------------------------------------------------------------------------
// Weight pre-pass: wt[n,k] = (f16) w[k,n]  (N-major so a lane's 16 K-values
// for one output column are 32 contiguous bytes -> two global_load_b128).
// ---------------------------------------------------------------------------
__global__ void convert_wT_kernel(const float* __restrict__ w,
                                  _Float16* __restrict__ wt, int K, int N) {
    int idx = blockIdx.x * blockDim.x + threadIdx.x;
    if (idx >= K * N) return;
    int k = idx / N;
    int n = idx % N;
    wt[(size_t)n * K + k] = (_Float16)w[(size_t)k * N + n];
}

// ---------------------------------------------------------------------------
// Input embedding: out[b,l,n] = x[b,l,:]@in_w[:,n] + in_b[n] + PE[l,n]
// ---------------------------------------------------------------------------
__global__ void embed_kernel(const float* __restrict__ x,
                             const float* __restrict__ w,
                             const float* __restrict__ b,
                             float* __restrict__ out, int L) {
    int idx = blockIdx.x * blockDim.x + threadIdx.x;
    int total = BATCH * L * DMODEL;
    if (idx >= total) return;
    int n  = idx % DMODEL;
    int bl = idx / DMODEL;
    int l  = bl % L;
    const float* xr = x + (size_t)bl * 32;
    float acc = b[n];
#pragma unroll
    for (int k = 0; k < 32; ++k) acc += xr[k] * w[k * DMODEL + n];
    int   i2  = n & ~1;
    float div = expf((float)i2 * (-9.210340371976184f / (float)DMODEL)); // ln(1e4)
    float ang = (float)l * div;
    acc += (n & 1) ? cosf(ang) : sinf(ang);
    out[idx] = acc;
}

// ---------------------------------------------------------------------------
// LayerNorm over last dim (512). One 256-thread block per row.
// ---------------------------------------------------------------------------
__global__ void layernorm_kernel(const float* __restrict__ x,
                                 const float* __restrict__ g,
                                 const float* __restrict__ beta,
                                 float* __restrict__ y, int rows) {
    int row = blockIdx.x;
    if (row >= rows) return;
    const float* xr = x + (size_t)row * DMODEL;
    float*       yr = y + (size_t)row * DMODEL;
    int t = threadIdx.x;
    float v0 = xr[t], v1 = xr[t + 256];
    __shared__ float red[256];
    red[t] = v0 + v1;
    __syncthreads();
    for (int s = 128; s > 0; s >>= 1) { if (t < s) red[t] += red[t + s]; __syncthreads(); }
    float mu = red[0] * (1.0f / DMODEL);
    __syncthreads();
    float d0 = v0 - mu, d1 = v1 - mu;
    red[t] = d0 * d0 + d1 * d1;
    __syncthreads();
    for (int s = 128; s > 0; s >>= 1) { if (t < s) red[t] += red[t + s]; __syncthreads(); }
    float rstd = rsqrtf(red[0] * (1.0f / DMODEL) + 1e-5f);
    yr[t]       = d0 * rstd * g[t]       + beta[t];
    yr[t + 256] = d1 * rstd * g[t + 256] + beta[t + 256];
}

// ---------------------------------------------------------------------------
// WMMA GEMM: out[M,N] = act(A[M,K] @ W + bias) (+ res), W given as f16 [N,K].
// 4-wave workgroup computes a 16x64 strip: the 16x64 A panel is staged to
// LDS once per K-step (async-to-LDS when available, ASYNCcnt-tracked) and
// shared by all 4 waves; each wave owns one 16x16 N-tile, 2 WMMAs / K-step.
//
// AMODE==0: A row-major [M,K].
// AMODE==1: A is ctx [B,H,Lrow,DH]; 8-float chunks start at multiples of 8
//           so they never cross a DH=64 head block.
//
// Fragment layouts used here (contiguous K order, same for A and B):
//   A (16x32 f16): lane m=lane&15; hi lanes K-offset 16; half j -> K=j+16*hi
//                  (two contiguous 8-chunks -> ds_load_b128).
//   B (32x16 f16): lane n=lane&15; hi lanes K-offset 16; half j -> K=j+koff
//                  (16 contiguous halves in N-major wt -> 2x global b128).
//   C/D (16x16 f32): VGPR r -> row r + 8*(lane>=16), col = lane&15.
// ---------------------------------------------------------------------------
template <int AMODE>
__global__ __launch_bounds__(128)
void gemm_wmma_kernel(const float* __restrict__ A,
                      const _Float16* __restrict__ Wt,
                      const float* __restrict__ bias,
                      const float* __restrict__ res,
                      float* __restrict__ out,
                      int M, int N, int K, int Lrow, int act) {
    __shared__ float tileA[16 * 64];          // 4 KB A panel (16 rows x 64 K)

    int tid   = threadIdx.x;                  // 0..127 (4 waves)
    int wave  = tid >> 5;
    int lane  = tid & 31;
    int tileM = blockIdx.y * 16;
    int tileN = blockIdx.x * 64 + wave * 16;
    int ml    = lane & 15;
    int hi    = lane >> 4;
    int abase = hi * 8;
    int koffB = hi * 16;
    int colB  = tileN + ml;

    // Staging role: 1024 floats / 128 threads = 8 floats (two b128) each.
    int srow = tid >> 3;                      // 0..15 : panel row
    int sseg = tid & 7;                       // 0..7  : 8-float segment
    int growA = tileM + srow;
    const float* gbase;
    size_t hstride = 0;
    if (AMODE == 0) {
        gbase = A + (size_t)growA * K;
    } else {
        int bb = growA / Lrow;
        int ll = growA - bb * Lrow;
        gbase   = A + ((size_t)bb * NHEADS * Lrow + ll) * DH;
        hstride = (size_t)Lrow * DH;
    }
    float* ldst = &tileA[srow * 64 + sseg * 8];
    const _Float16* wcol = Wt + (size_t)colB * K + koffB;

    v8f acc = {};
    for (int kk = 0; kk < K; kk += 64) {
        // ---- stage A panel: global -> LDS ----
        int kg = kk + sseg * 8;
        const float* gp = (AMODE == 0)
            ? (gbase + kg)
            : (gbase + (size_t)(kg >> 6) * hstride + (kg & 63));
#if HAS_ASYNC_LDS
        __builtin_amdgcn_global_load_async_to_lds_b128(
            (gptr4)(v4i*)const_cast<float*>(gp),
            (sptr4)(v4i*)ldst, 0, 0);
        __builtin_amdgcn_global_load_async_to_lds_b128(
            (gptr4)(v4i*)const_cast<float*>(gp + 4),
            (sptr4)(v4i*)(ldst + 4), 0, 0);
        __builtin_amdgcn_s_wait_asynccnt(0);
#else
        v4f t0 = *(const v4f*)gp;
        v4f t1 = *(const v4f*)(gp + 4);
        *(v4f*)ldst = t0;
        *(v4f*)(ldst + 4) = t1;
#endif
        __syncthreads();

        // ---- consume: 2 x (A frag from LDS, B frag from global, WMMA) ----
#pragma unroll
        for (int s = 0; s < 2; ++s) {
            int k0 = 32 * s + 2 * abase;      // chunk 0 within panel
            int k1 = k0 + 8;                  // chunk 1 within panel
            v4f a0 = *(const v4f*)&tileA[ml * 64 + k0];
            v4f a1 = *(const v4f*)&tileA[ml * 64 + k0 + 4];
            v4f a2 = *(const v4f*)&tileA[ml * 64 + k1];
            v4f a3 = *(const v4f*)&tileA[ml * 64 + k1 + 4];
            v16h a;
#pragma unroll
            for (int j = 0; j < 4; ++j) {
                a[j]      = (_Float16)a0[j];
                a[j + 4]  = (_Float16)a1[j];
                a[j + 8]  = (_Float16)a2[j];
                a[j + 12] = (_Float16)a3[j];
            }
            v16h bf = *(const v16h*)(wcol + kk + 32 * s);   // 32B -> 2x b128
            acc = __builtin_amdgcn_wmma_f32_16x16x32_f16(false, a, false, bf,
                                                         (short)0, acc, false, false);
        }
        if (kk + 64 < K) __builtin_prefetch(wcol + kk + 64, 0, 0);  // global_prefetch_b8
        __syncthreads();                      // panel consumed by all waves
    }
#pragma unroll
    for (int r = 0; r < 8; ++r) {
        int row = tileM + abase + r;
        float v = acc[r] + bias[colB];
        if (act == 1) v = gelu_exact(v);
        size_t o = (size_t)row * N + colB;
        if (res) v += res[o];
        out[o] = v;
    }
}

// ---------------------------------------------------------------------------
// ProbSparse stats: M[b,h,q] = max_k(s) - mean_k(s), s = q.k / 8.
// ---------------------------------------------------------------------------
__global__ void score_stats_kernel(const float* __restrict__ Q,
                                   const float* __restrict__ Kb,
                                   float* __restrict__ Mout, int L) {
    int idx = blockIdx.x * blockDim.x + threadIdx.x;
    int total = BATCH * NHEADS * L;
    if (idx >= total) return;
    int q  = idx % L;
    int bh = idx / L;
    int h  = bh % NHEADS;
    int b  = bh / NHEADS;
    const float* qr = Q + ((size_t)(b * L + q)) * DMODEL + h * DH;
    float qv[DH];
#pragma unroll
    for (int d = 0; d < DH; ++d) qv[d] = qr[d];
    float mx = -__builtin_inff();
    float sum = 0.0f;
    for (int k = 0; k < L; ++k) {
        const float* kr = Kb + ((size_t)(b * L + k)) * DMODEL + h * DH;
        float dot = 0.0f;
#pragma unroll
        for (int d = 0; d < DH; ++d) dot += qv[d] * kr[d];
        dot *= 0.125f;
        mx = fmaxf(mx, dot);
        sum += dot;
    }
    Mout[idx] = mx - sum / (float)L;
}

// Top-u per (b,h): selection-max, strictly-greater keeps lowest index on ties
// (matches jax.lax.top_k). Destroys Mbuf (dead afterwards).
__global__ void topk_kernel(float* __restrict__ Mbuf, int* __restrict__ idx,
                            int L, int u) {
    int bh = blockIdx.x * blockDim.x + threadIdx.x;
    if (bh >= BATCH * NHEADS) return;
    float* mrow = Mbuf + (size_t)bh * L;
    int*   irow = idx + bh * u;
    for (int t = 0; t < u; ++t) {
        float best = -__builtin_inff();
        int   bi   = 0;
        for (int k = 0; k < L; ++k) {
            float v = mrow[k];
            if (v > best) { best = v; bi = k; }
        }
        irow[t] = bi;
        mrow[bi] = -__builtin_inff();
    }
}

// sel[b,h,i,k] = q[idx[i]] . k / 8
__global__ void sel_scores_kernel(const float* __restrict__ Q,
                                  const float* __restrict__ Kb,
                                  const int* __restrict__ idx,
                                  float* __restrict__ sel, int L, int u) {
    long long t = (long long)blockIdx.x * blockDim.x + threadIdx.x;
    long long total = (long long)BATCH * NHEADS * u * L;
    if (t >= total) return;
    int k  = (int)(t % L);
    long long r = t / L;
    int i  = (int)(r % u);
    int bh = (int)(r / u);
    int h  = bh % NHEADS;
    int b  = bh / NHEADS;
    int q  = idx[bh * u + i];
    const float* qr = Q  + ((size_t)(b * L + q)) * DMODEL + h * DH;
    const float* kr = Kb + ((size_t)(b * L + k)) * DMODEL + h * DH;
    float dot = 0.0f;
#pragma unroll
    for (int d = 0; d < DH; ++d) dot += qr[d] * kr[d];
    sel[t] = dot * 0.125f;
}

// Row softmax over length L (<=512); one 256-thread block per row.
__global__ void softmax_kernel(float* __restrict__ sel, int L) {
    int row = blockIdx.x;
    float* sr = sel + (size_t)row * L;
    int t = threadIdx.x;
    __shared__ float red[256];
    float v0 = (t < L)       ? sr[t]       : -__builtin_inff();
    float v1 = (t + 256 < L) ? sr[t + 256] : -__builtin_inff();
    red[t] = fmaxf(v0, v1);
    __syncthreads();
    for (int s = 128; s > 0; s >>= 1) { if (t < s) red[t] = fmaxf(red[t], red[t + s]); __syncthreads(); }
    float mx = red[0];
    __syncthreads();
    float e0 = (t < L)       ? expf(v0 - mx) : 0.0f;
    float e1 = (t + 256 < L) ? expf(v1 - mx) : 0.0f;
    red[t] = e0 + e1;
    __syncthreads();
    for (int s = 128; s > 0; s >>= 1) { if (t < s) red[t] += red[t + s]; __syncthreads(); }
    float inv = 1.0f / red[0];
    if (t < L)       sr[t]       = e0 * inv;
    if (t + 256 < L) sr[t + 256] = e1 * inv;
}

__global__ void meanv_kernel(const float* __restrict__ V,
                             float* __restrict__ meanv, int L) {
    int idx = blockIdx.x * blockDim.x + threadIdx.x;
    if (idx >= BATCH * NHEADS * DH) return;
    int d  = idx % DH;
    int bh = idx / DH;
    int h  = bh % NHEADS;
    int b  = bh / NHEADS;
    float s = 0.0f;
    for (int l = 0; l < L; ++l)
        s += V[((size_t)(b * L + l)) * DMODEL + h * DH + d];
    meanv[idx] = s / (float)L;
}

// ctx[b,h,l,d] = meanv[b,h,d]
__global__ void ctx_fill_kernel(const float* __restrict__ meanv,
                                float* __restrict__ ctx, int L) {
    int idx = blockIdx.x * blockDim.x + threadIdx.x;
    int total = BATCH * NHEADS * L * DH;
    if (idx >= total) return;
    int d  = idx % DH;
    int r  = idx / DH;
    int bh = r / L;
    ctx[idx] = meanv[bh * DH + d];
}

// ctx[b,h,idx[i],d] = sum_k sel_p[b,h,i,k] * V[b,k,h,d]
__global__ void attn_scatter_kernel(const float* __restrict__ sel,
                                    const float* __restrict__ V,
                                    const int* __restrict__ idx,
                                    float* __restrict__ ctx, int L, int u) {
    int t = blockIdx.x * blockDim.x + threadIdx.x;
    int total = BATCH * NHEADS * u * DH;
    if (t >= total) return;
    int d  = t % DH;
    int r  = t / DH;
    int i  = r % u;
    int bh = r / u;
    int h  = bh % NHEADS;
    int b  = bh / NHEADS;
    const float* p = sel + ((size_t)bh * u + i) * L;
    float acc = 0.0f;
    for (int k = 0; k < L; ++k)
        acc += p[k] * V[((size_t)(b * L + k)) * DMODEL + h * DH + d];
    int q = idx[bh * u + i];
    ctx[(((size_t)bh) * L + q) * DH + d] = acc;
}

// conv1d(k=3, pad=1) over length + maxpool2: out[b,l2,d]
__global__ void conv_pool_kernel(const float* __restrict__ x,
                                 const float* __restrict__ w,
                                 const float* __restrict__ b,
                                 float* __restrict__ out, int L) {
    int idx = blockIdx.x * blockDim.x + threadIdx.x;
    int Lh = L >> 1;
    int total = BATCH * Lh * DMODEL;
    if (idx >= total) return;
    int d  = idx % DMODEL;
    int r  = idx / DMODEL;
    int l2 = r % Lh;
    int bb = r / Lh;
    float y[2];
#pragma unroll
    for (int j = 0; j < 2; ++j) {
        int l = 2 * l2 + j;
        float acc = b[d];
        const float* xm = x + ((size_t)bb * L) * DMODEL;
        for (int c = 0; c < DMODEL; ++c) {
            const float* wr = w + ((size_t)d * DMODEL + c) * 3;
            float x0 = (l - 1 >= 0) ? xm[(size_t)(l - 1) * DMODEL + c] : 0.0f;
            float x1 = xm[(size_t)l * DMODEL + c];
            float x2 = (l + 1 < L)  ? xm[(size_t)(l + 1) * DMODEL + c] : 0.0f;
            acc += x0 * wr[0] + x1 * wr[1] + x2 * wr[2];
        }
        y[j] = acc;
    }
    out[idx] = fmaxf(y[0], y[1]);
}

__global__ void meanpool_kernel(const float* __restrict__ x,
                                float* __restrict__ out, int L) {
    int idx = blockIdx.x * blockDim.x + threadIdx.x;
    if (idx >= BATCH * DMODEL) return;
    int d = idx % DMODEL;
    int b = idx / DMODEL;
    float s = 0.0f;
    for (int l = 0; l < L; ++l) s += x[((size_t)(b * L + l)) * DMODEL + d];
    out[idx] = s / (float)L;
}

// classifier tail: out[b,n] = h[b,:]@w[:,n] + bias[n], N=2, K=256
__global__ void cls2_kernel(const float* __restrict__ h,
                            const float* __restrict__ w,
                            const float* __restrict__ b,
                            float* __restrict__ out) {
    int idx = blockIdx.x * blockDim.x + threadIdx.x;
    if (idx >= BATCH * 2) return;
    int n  = idx & 1;
    int bb = idx >> 1;
    float acc = b[n];
    for (int k = 0; k < 256; ++k) acc += h[bb * 256 + k] * w[k * 2 + n];
    out[idx] = acc;
}

// ---------------------------------------------------------------------------
// Host orchestration
// ---------------------------------------------------------------------------
extern "C" void kernel_launch(void* const* d_in, const int* in_sizes, int n_in,
                              void* d_out, int out_size, void* d_ws, size_t ws_size,
                              hipStream_t stream) {
    (void)in_sizes; (void)n_in; (void)out_size; (void)ws_size;
    auto P = [&](int i) -> const float* { return (const float*)d_in[i]; };

    // Leaf order: insertion order of setup_inputs() dict (79 leaves).
    const float* x_in = P(0);
    const float* in_w = P(1);
    const float* in_b = P(2);
    const float* conv_w[3]; const float* conv_b[3];
    for (int i = 0; i < 3; ++i) { conv_w[i] = P(67 + i); conv_b[i] = P(70 + i); }
    const float* norm_g = P(73); const float* norm_b = P(74);
    const float* cls_w1 = P(75); const float* cls_b1 = P(76);
    const float* cls_w2 = P(77); const float* cls_b2 = P(78);

    // Workspace layout (f32 slots unless noted). Shared region overlaps
    // Q/K/V/attn scratch with the FF hidden buffer (disjoint lifetimes).
    const size_t XSZ = (size_t)BATCH * 512 * DMODEL;   // 16,777,216 floats
    const size_t FSZ = (size_t)BATCH * 512 * DFF;      // 67,108,864 floats
    float* ws   = (float*)d_ws;
    float* xbuf = ws;
    float* sbuf = ws + XSZ;
    float* sh   = ws + 2 * XSZ;
    float* Q    = sh;
    float* Kb   = sh + XSZ;
    float* V    = sh + 2 * XSZ;
    float* Mrow = sh + 3 * XSZ;
    float* meanv = Mrow + (size_t)BATCH * NHEADS * 512;
    float* sel   = meanv + (size_t)BATCH * NHEADS * DH;
    int*   idx   = (int*)(sel + (size_t)BATCH * NHEADS * 35 * 512);
    float* fbuf  = sh;                                  // FF hidden (overlap)

    // f16 transposed weights past the shared region (persist whole launch).
    _Float16* wt = (_Float16*)(ws + 2 * XSZ + FSZ);
    const size_t WSQ = (size_t)DMODEL * DMODEL;         // 262,144
    const size_t WS1 = (size_t)DMODEL * DFF;            // 1,048,576
    _Float16 *wqT[4], *wkT[4], *wvT[4], *woT[4], *w1T[4], *w2T[4];
    _Float16* p = wt;
    for (int i = 0; i < 4; ++i) {
        wqT[i] = p; p += WSQ;  wkT[i] = p; p += WSQ;
        wvT[i] = p; p += WSQ;  woT[i] = p; p += WSQ;
        w1T[i] = p; p += WS1;  w2T[i] = p; p += WS1;
    }
    _Float16* c1T = p;                                   // cls_w1 [256,512] f16

    // Weight conversion pre-pass.
    int gsq = (int)((WSQ + 255) / 256), gs1 = (int)((WS1 + 255) / 256);
    for (int i = 0; i < 4; ++i) {
        int base = 3 + i * 16;
        convert_wT_kernel<<<gsq, 256, 0, stream>>>(P(base + 0),  wqT[i], DMODEL, DMODEL);
        convert_wT_kernel<<<gsq, 256, 0, stream>>>(P(base + 2),  wkT[i], DMODEL, DMODEL);
        convert_wT_kernel<<<gsq, 256, 0, stream>>>(P(base + 4),  wvT[i], DMODEL, DMODEL);
        convert_wT_kernel<<<gsq, 256, 0, stream>>>(P(base + 6),  woT[i], DMODEL, DMODEL);
        convert_wT_kernel<<<gs1, 256, 0, stream>>>(P(base + 8),  w1T[i], DMODEL, DFF);
        convert_wT_kernel<<<gs1, 256, 0, stream>>>(P(base + 10), w2T[i], DFF, DMODEL);
    }
    convert_wT_kernel<<<(DMODEL * 256 + 255) / 256, 256, 0, stream>>>(cls_w1, c1T, DMODEL, 256);

    int L = 512;
    {
        int total = BATCH * L * DMODEL;
        embed_kernel<<<(total + 255) / 256, 256, 0, stream>>>(x_in, in_w, in_b, xbuf, L);
    }

    for (int layer = 0; layer < 4; ++layer) {
        int base = 3 + layer * 16;
        const float *bq = P(base + 1), *bk = P(base + 3), *bv = P(base + 5), *bo = P(base + 7);
        const float *b1 = P(base + 9), *b2 = P(base + 11);
        const float *n1g = P(base + 12), *n1b = P(base + 13);
        const float *n2g = P(base + 14), *n2b = P(base + 15);

        int M = BATCH * L;
        // --- attention block ---
        layernorm_kernel<<<M, 256, 0, stream>>>(xbuf, n1g, n1b, sbuf, M);

        dim3 gqkv(DMODEL / 64, M / 16);
        gemm_wmma_kernel<0><<<gqkv, 128, 0, stream>>>(sbuf, wqT[layer], bq, nullptr, Q,  M, DMODEL, DMODEL, L, 0);
        gemm_wmma_kernel<0><<<gqkv, 128, 0, stream>>>(sbuf, wkT[layer], bk, nullptr, Kb, M, DMODEL, DMODEL, L, 0);
        gemm_wmma_kernel<0><<<gqkv, 128, 0, stream>>>(sbuf, wvT[layer], bv, nullptr, V,  M, DMODEL, DMODEL, L, 0);

        int u = (int)(5.0 * ceil(log((double)L)));
        if (u > L) u = L;

        int tstat = BATCH * NHEADS * L;
        score_stats_kernel<<<(tstat + 255) / 256, 256, 0, stream>>>(Q, Kb, Mrow, L);
        topk_kernel<<<(BATCH * NHEADS + 63) / 64, 64, 0, stream>>>(Mrow, idx, L, u);

        long long tsel = (long long)BATCH * NHEADS * u * L;
        sel_scores_kernel<<<(unsigned)((tsel + 255) / 256), 256, 0, stream>>>(Q, Kb, idx, sel, L, u);
        softmax_kernel<<<BATCH * NHEADS * u, 256, 0, stream>>>(sel, L);

        meanv_kernel<<<(BATCH * NHEADS * DH + 255) / 256, 256, 0, stream>>>(V, meanv, L);
        int tctx = BATCH * NHEADS * L * DH;
        ctx_fill_kernel<<<(tctx + 255) / 256, 256, 0, stream>>>(meanv, sbuf, L);
        int tsc = BATCH * NHEADS * u * DH;
        attn_scatter_kernel<<<(tsc + 255) / 256, 256, 0, stream>>>(sel, V, idx, sbuf, L, u);

        // out projection (ctx gather mode) + residual into xbuf
        gemm_wmma_kernel<1><<<gqkv, 128, 0, stream>>>(sbuf, woT[layer], bo, xbuf, xbuf, M, DMODEL, DMODEL, L, 0);

        // --- feed-forward block ---
        layernorm_kernel<<<M, 256, 0, stream>>>(xbuf, n2g, n2b, sbuf, M);
        dim3 g1(DFF / 64, M / 16);
        gemm_wmma_kernel<0><<<g1, 128, 0, stream>>>(sbuf, w1T[layer], b1, nullptr, fbuf, M, DFF, DMODEL, L, 1);
        dim3 g2(DMODEL / 64, M / 16);
        gemm_wmma_kernel<0><<<g2, 128, 0, stream>>>(fbuf, w2T[layer], b2, xbuf, xbuf, M, DMODEL, DFF, L, 0);

        // --- conv + maxpool distillation ---
        if (layer < 3) {
            int Lh = L / 2;
            int tc = BATCH * Lh * DMODEL;
            conv_pool_kernel<<<(tc + 255) / 256, 256, 0, stream>>>(xbuf, conv_w[layer], conv_b[layer], sbuf, L);
            float* t = xbuf; xbuf = sbuf; sbuf = t;
            L = Lh;
        }
    }

    // final norm -> mean pool -> classifier
    int M = BATCH * L;   // L == 64
    layernorm_kernel<<<M, 256, 0, stream>>>(xbuf, norm_g, norm_b, sbuf, M);
    float* pooled = Q;
    float* clsh   = Q + (size_t)BATCH * DMODEL;
    meanpool_kernel<<<(BATCH * DMODEL + 255) / 256, 256, 0, stream>>>(sbuf, pooled, L);
    dim3 gc(256 / 64, BATCH / 16);
    gemm_wmma_kernel<0><<<gc, 128, 0, stream>>>(pooled, c1T, cls_b1, nullptr, clsh, BATCH, 256, DMODEL, L, 1);
    cls2_kernel<<<1, 128, 0, stream>>>(clsh, cls_w2, cls_b2, (float*)d_out);
}
